// TopoFuSAGNet_51891794870336
// MI455X (gfx1250) — matrix operand order repaired and checked by reading, not compiled, not run on hardware
//
#include <hip/hip_runtime.h>
#include <hip/hip_bf16.h>
#include <math.h>

// ---------------- problem constants ----------------
#define BB 32
#define NN 512
#define WW 256
#define CC 16
#define ZZ 8
#define EE 8
#define HD 16
#define GOUT 32
#define ROWS (BB * NN)          // 16384
#define MAXNBR 24               // top-20 + diag <= 21

// output layout (float32): predicted[16384], recon[4194304], kl[1], sdev[16384]
#define OUT_PRED 0
#define OUT_RECON 16384
#define OUT_KL   (16384 + 4194304)
#define OUT_SDEV (OUT_KL + 1)

typedef __attribute__((ext_vector_type(16))) _Float16 v16h;
typedef __attribute__((ext_vector_type(8)))  float    v8f;
typedef __attribute__((ext_vector_type(4)))  float    f4v;

union Frag16 { v16h v; f4v q[2]; };   // 32 bytes: one WMMA f16 A/B operand

static __device__ __forceinline__ f4v ld4(const float* p) {
  return *reinterpret_cast<const f4v*>(p);
}
static __device__ __forceinline__ f4v ld4h(const _Float16* p) {
  return *reinterpret_cast<const f4v*>(p);
}

// =====================================================================
// 1) MSTCN + time-mean, collapsed: mean_t conv_k(x) = b + (1/W)*sum_j w_j*(S - corr)
//    One wave per (b,n) row; wave32 butterfly reduction for S; edge sums via shfl.
// =====================================================================
__global__ void k_mstcn(const float* __restrict__ x,
                        const float* w2, const float* b2,
                        const float* w3, const float* b3,
                        const float* w5, const float* b5,
                        const float* w7, const float* b7,
                        float* __restrict__ nf) {
  const int wv = threadIdx.x >> 5, lane = threadIdx.x & 31;
  const int row = blockIdx.x * 8 + wv;                 // 2048 blocks * 8 waves
  const float* xr = x + (size_t)row * WW;
  f4v xa = ld4(xr + lane * 8);
  f4v xb = ld4(xr + lane * 8 + 4);
  float s = xa.x + xa.y + xa.z + xa.w + xb.x + xb.y + xb.z + xb.w;
  #pragma unroll
  for (int off = 16; off; off >>= 1) s += __shfl_xor(s, off, 32);
  // prefix sums of x[0..2] (lane 0) and suffix sums of x[253..255] (lane 31)
  float p1 = xa.x, p2 = p1 + xa.y, p3 = p2 + xa.z;
  float q1 = xb.w, q2 = q1 + xb.z, q3 = q2 + xb.y;
  p1 = __shfl(p1, 0, 32);  p2 = __shfl(p2, 0, 32);  p3 = __shfl(p3, 0, 32);
  q1 = __shfl(q1, 31, 32); q2 = __shfl(q2, 31, 32); q3 = __shfl(q3, 31, 32);
  const float pre[4] = {0.f, p1, p2, p3};
  const float suf[4] = {0.f, q1, q2, q3};
  const float* wp[4] = {w2, w3, w5, w7};
  const float* bp[4] = {b2, b3, b5, b7};
  const int ks[4] = {2, 3, 5, 7};
  for (int o = lane; o < 4 * CC; o += 32) {
    const int br = o >> 4, c = o & 15;
    const int k = ks[br], pl = (k - 1) >> 1;
    const float* w = wp[br] + c * k;
    float acc = 0.f;
    for (int j = 0; j < k; ++j) {
      const int d = j - pl;
      const float corr = (d > 0) ? pre[d] : ((d < 0) ? suf[-d] : 0.f);
      acc += w[j] * (s - corr);
    }
    nf[(size_t)row * 64 + o] = acc * (1.f / (float)WW) + bp[br][c];
  }
}

// =====================================================================
// 2) SAE encode: z = nf @ enc_w + enc_b; sigmoid stats (deterministic block
//    partials), sparsity_dev output, and f16 gat_in = [z | emb | 0pad16].
// =====================================================================
__global__ void k_encode(const float* __restrict__ nf,
                         const float* __restrict__ enc_w, const float* __restrict__ enc_b,
                         const float* __restrict__ semb,
                         float* __restrict__ z, _Float16* __restrict__ gi,
                         float* __restrict__ sdev, float* __restrict__ rho_part) {
  __shared__ float sh[256][ZZ];
  const int tid = threadIdx.x;
  const int row = blockIdx.x * 256 + tid;              // 64 blocks
  const float* f = nf + (size_t)row * 64;
  float zz[ZZ];
  #pragma unroll
  for (int k = 0; k < ZZ; ++k) zz[k] = enc_b[k];
  for (int d = 0; d < 64; ++d) {
    const float fv = f[d];
    #pragma unroll
    for (int k = 0; k < ZZ; ++k) zz[k] += fv * enc_w[d * ZZ + k];
  }
  float dev = 0.f;
  #pragma unroll
  for (int k = 0; k < ZZ; ++k) {
    const float a = 1.f / (1.f + __expf(-zz[k]));
    sh[tid][k] = a;
    dev += fabsf(a - 0.05f);
    z[(size_t)row * ZZ + k] = zz[k];
    gi[(size_t)row * 32 + k] = (_Float16)zz[k];
  }
  const int n = row & (NN - 1);
  #pragma unroll
  for (int k = 0; k < EE; ++k) gi[(size_t)row * 32 + 8 + k] = (_Float16)semb[n * EE + k];
  #pragma unroll
  for (int k = 0; k < 16; ++k) gi[(size_t)row * 32 + 16 + k] = (_Float16)0.f;
  sdev[row] = dev * (1.f / (float)ZZ);
  __syncthreads();
  if (tid < ZZ) {                                      // deterministic serial sum
    float ssum = 0.f;
    for (int i = 0; i < 256; ++i) ssum += sh[i][tid];
    rho_part[blockIdx.x * ZZ + tid] = ssum;
  }
}

__global__ void k_kl(const float* __restrict__ rho_part, float* __restrict__ kl_out) {
  if (threadIdx.x == 0) {
    float kl = 0.f;
    for (int k = 0; k < ZZ; ++k) {
      float s = 0.f;
      for (int b = 0; b < 64; ++b) s += rho_part[b * ZZ + k];
      float rh = s * (1.f / (float)ROWS);
      rh = fminf(fmaxf(rh, 1e-6f), 1.f - 1e-6f);
      kl += 0.05f * logf(0.05f / rh) + 0.95f * logf(0.95f / (1.f - rh));
    }
    kl_out[0] = kl;
  }
}

// =====================================================================
// 3) recon = z @ dec_w + dec_b  (K=8: bandwidth-bound 16.8MB write)
// =====================================================================
__global__ void k_recon(const float* __restrict__ z, const float* __restrict__ dec_w,
                        const float* __restrict__ dec_b, float* __restrict__ recon) {
  __shared__ float zz[ZZ];
  const int row = blockIdx.x, col = threadIdx.x;       // 16384 x 256
  if (col < ZZ) zz[col] = z[(size_t)row * ZZ + col];
  __syncthreads();
  float acc = dec_b[col];
  #pragma unroll
  for (int k = 0; k < ZZ; ++k) acc += zz[k] * dec_w[k * WW + col];
  recon[(size_t)row * WW + col] = acc;
}

// =====================================================================
// 4) weight prep: f16 transposed [N][K] layouts for WMMA B-fragments
//    WcatT: K<16 -> [gat0_w | gat1_w], K>=16 zero-padded.  projT: proj_w^T.
// =====================================================================
__global__ void k_prep(const float* g0w, const float* g1w, const float* pw,
                       _Float16* __restrict__ WcatT, _Float16* __restrict__ projT) {
  const int t = blockIdx.x * blockDim.x + threadIdx.x; // 1024 threads
  if (t >= 1024) return;
  const int n = t >> 5, k = t & 31;
  float wv = 0.f;
  if (k < 16) wv = (n < 16) ? g0w[k * 16 + n] : g1w[k * 16 + (n - 16)];
  WcatT[n * 32 + k] = (_Float16)wv;
  projT[n * 32 + k] = (_Float16)pw[k * 32 + n];
}

// =====================================================================
// 5) graph learning: cosine sim, top-20 per row (wave32 argmax x20),
//    adjacency kept as sparse neighbor lists (A only gates via a>0).
// =====================================================================
__global__ void k_graph(const float* __restrict__ emb,
                        int* __restrict__ nbr, int* __restrict__ cnt) {
  __shared__ int flags[NN];
  const int i = blockIdx.x, lane = threadIdx.x;        // 512 blocks x 32
  float ei[EE];
  #pragma unroll
  for (int d = 0; d < EE; ++d) ei[d] = emb[i * EE + d];
  float ni = 0.f;
  #pragma unroll
  for (int d = 0; d < EE; ++d) ni += ei[d] * ei[d];
  ni = fmaxf(sqrtf(ni), 1e-12f);
  float vals[16];
  for (int t = 0; t < 16; ++t) {
    const int j = lane + 32 * t;
    float dot = 0.f, nj = 0.f;
    #pragma unroll
    for (int d = 0; d < EE; ++d) { const float e = emb[j * EE + d]; dot += ei[d] * e; nj += e * e; }
    nj = fmaxf(sqrtf(nj), 1e-12f);
    vals[t] = dot / (ni * nj) * 0.3535533906f;         // / sqrt(E)
  }
  unsigned selmask = 0;
  for (int it = 0; it < 20; ++it) {
    float bv = -1e30f; int bj = 1 << 30;
    for (int t = 0; t < 16; ++t) {
      const int j = lane + 32 * t;
      if (!((selmask >> t) & 1) && j != i) {
        if (vals[t] > bv || (vals[t] == bv && j < bj)) { bv = vals[t]; bj = j; }
      }
    }
    #pragma unroll
    for (int off = 16; off; off >>= 1) {
      const float ov = __shfl_xor(bv, off, 32);
      const int   oj = __shfl_xor(bj, off, 32);
      if (ov > bv || (ov == bv && oj < bj)) { bv = ov; bj = oj; }
    }
    if ((bj & 31) == lane) selmask |= 1u << (bj >> 5);
  }
  for (int t = 0; t < 16; ++t) {
    const int j = lane + 32 * t;
    flags[j] = (((selmask >> t) & 1) && vals[t] > 0.f) || (j == i);
  }
  __syncthreads();
  if (lane == 0) {
    int c = 0;
    for (int j = 0; j < NN; ++j) if (flags[j]) nbr[i * MAXNBR + (c++)] = j;
    cnt[i] = c;
  }
}

// =====================================================================
// 6) h = gat_in @ [W0|W1] via V_WMMA_F32_16X16X32_F16 (K=16 zero-padded to 32)
//    one wave per 16-row tile; two N-tiles (head0 cols 0-15, head1 cols 16-31)
// =====================================================================
__global__ void k_gat_h(const _Float16* __restrict__ gi,
                        const _Float16* __restrict__ WcatT,
                        float* __restrict__ h) {
  const int lane = threadIdx.x & 31;
  const int tile = (blockIdx.x * blockDim.x + threadIdx.x) >> 5;  // 1024 tiles
  const int r = lane & 15, half = lane >> 4;
  // A fragment: row = lane%16; K chunk = (lane/16)*16 (contiguous, matches B)
  Frag16 a;
  const _Float16* ap = gi + (size_t)(tile * 16 + r) * 32;
  a.q[0] = ld4h(ap + half * 16);
  a.q[1] = ld4h(ap + half * 16 + 8);
  // B fragments from [N][K] transposed weights: col = lane%16, K chunk = (lane/16)*16
  const int kb = half * 16;
  Frag16 b0, b1;
  b0.q[0] = ld4h(WcatT + r * 32 + kb);      b0.q[1] = ld4h(WcatT + r * 32 + kb + 8);
  b1.q[0] = ld4h(WcatT + (16 + r) * 32 + kb); b1.q[1] = ld4h(WcatT + (16 + r) * 32 + kb + 8);
  v8f c0 = {}, c1 = {};
  c0 = __builtin_amdgcn_wmma_f32_16x16x32_f16(false, a.v, false, b0.v, (short)0, c0, false, false);
  c1 = __builtin_amdgcn_wmma_f32_16x16x32_f16(false, a.v, false, b1.v, (short)0, c1, false, false);
  // C layout: lane holds N = lane%16, rows M = rr + 8*(lane/16)
  const int orow = tile * 16;
  #pragma unroll
  for (int rr = 0; rr < 8; ++rr) {
    const int m = rr + 8 * half;
    h[(size_t)(orow + m) * 32 + r]      = c0[rr];
    h[(size_t)(orow + m) * 32 + 16 + r] = c1[rr];
  }
}

// =====================================================================
// 7) attention logits per node: e_src/e_dst for both heads
// =====================================================================
__global__ void k_edges(const float* __restrict__ h,
                        const float* a0s, const float* a0d,
                        const float* a1s, const float* a1d,
                        float* __restrict__ es, float* __restrict__ ed) {
  const int row = blockIdx.x * blockDim.x + threadIdx.x;  // 16384
  const float* hr = h + (size_t)row * 32;
  float e0s = 0.f, e0d = 0.f, e1s = 0.f, e1d = 0.f;
  #pragma unroll
  for (int d = 0; d < HD; ++d) {
    const float v0 = hr[d], v1 = hr[16 + d];
    e0s += v0 * a0s[d]; e0d += v0 * a0d[d];
    e1s += v1 * a1s[d]; e1d += v1 * a1d[d];
  }
  es[row] = e0s; ed[row] = e0d;
  es[ROWS + row] = e1s; ed[ROWS + row] = e1d;
}

// 8) masked softmax over <=21 neighbors -> normalized alpha
__global__ void k_stats(const float* __restrict__ es, const float* __restrict__ ed,
                        const int* __restrict__ nbr, const int* __restrict__ cnt,
                        float* __restrict__ alpha) {
  const int id = blockIdx.x * blockDim.x + threadIdx.x;   // 32768 = 2 heads * rows
  if (id >= 2 * ROWS) return;
  const int hh = id >> 14, row = id & (ROWS - 1);
  const int b = row >> 9, ni = row & (NN - 1);
  const float ei = es[hh * ROWS + row];
  const float* edh = ed + hh * ROWS + b * NN;
  const int c_ = cnt[ni];
  float ev[MAXNBR];
  float m = -1e30f;
  for (int c = 0; c < c_; ++c) {
    const int j = nbr[ni * MAXNBR + c];
    float e = ei + edh[j];
    e = (e > 0.f) ? e : 0.2f * e;                         // leaky_relu 0.2
    ev[c] = e; m = fmaxf(m, e);
  }
  float ssum = 0.f;
  for (int c = 0; c < c_; ++c) { const float w = __expf(ev[c] - m); ev[c] = w; ssum += w; }
  const float inv = 1.f / ssum;
  float* ap = alpha + (size_t)id * MAXNBR;
  for (int c = 0; c < c_; ++c) ap[c] = ev[c] * inv;
}

// 9) sparse alpha @ h  (<=21 nonzeros/row beats dense 512-K GEMM by 25x)
__global__ void k_aggr(const float* __restrict__ h, const float* __restrict__ alpha,
                       const int* __restrict__ nbr, const int* __restrict__ cnt,
                       _Float16* __restrict__ hp) {
  const int id = blockIdx.x * blockDim.x + threadIdx.x;   // 524288
  const int row = id >> 5, col = id & 31;
  const int hh = col >> 4, b = row >> 9, ni = row & (NN - 1);
  const int c_ = cnt[ni];
  const float* ap = alpha + ((size_t)hh * ROWS + row) * MAXNBR;
  const float* hb = h + (size_t)b * NN * 32 + col;
  float acc = 0.f;
  for (int c = 0; c < c_; ++c) {
    const int j = nbr[ni * MAXNBR + c];
    acc += ap[c] * hb[(size_t)j * 32];
  }
  hp[(size_t)row * 32 + col] = (_Float16)acc;
}

// =====================================================================
// 10) fused = elu(hp @ proj_w + b) via WMMA (K=32 exact); predicted = fused @ head_w
// =====================================================================
__global__ void k_proj(const _Float16* __restrict__ hp, const _Float16* __restrict__ projT,
                       const float* __restrict__ proj_b,
                       const float* __restrict__ head_w, const float* __restrict__ head_b,
                       float* __restrict__ pred) {
  __shared__ float tilebuf[8][16][32];
  const int wv = threadIdx.x >> 5, lane = threadIdx.x & 31;
  const int tile = blockIdx.x * 8 + wv;                  // 1024 tiles
  const int r = lane & 15, half = lane >> 4;
  Frag16 a;
  const _Float16* ap = hp + (size_t)(tile * 16 + r) * 32;
  a.q[0] = ld4h(ap + half * 16);
  a.q[1] = ld4h(ap + half * 16 + 8);
  const int kb = half * 16;
  Frag16 b0, b1;
  b0.q[0] = ld4h(projT + r * 32 + kb);        b0.q[1] = ld4h(projT + r * 32 + kb + 8);
  b1.q[0] = ld4h(projT + (16 + r) * 32 + kb); b1.q[1] = ld4h(projT + (16 + r) * 32 + kb + 8);
  v8f c0 = {}, c1 = {};
  c0 = __builtin_amdgcn_wmma_f32_16x16x32_f16(false, a.v, false, b0.v, (short)0, c0, false, false);
  c1 = __builtin_amdgcn_wmma_f32_16x16x32_f16(false, a.v, false, b1.v, (short)0, c1, false, false);
  #pragma unroll
  for (int rr = 0; rr < 8; ++rr) {
    const int m = rr + 8 * half;
    float v0 = c0[rr] + proj_b[r];
    float v1 = c1[rr] + proj_b[16 + r];
    v0 = (v0 > 0.f) ? v0 : (__expf(v0) - 1.f);           // ELU
    v1 = (v1 > 0.f) ? v1 : (__expf(v1) - 1.f);
    tilebuf[wv][m][r] = v0;
    tilebuf[wv][m][16 + r] = v1;
  }
  __syncthreads();
  if (lane < 16) {
    float acc = head_b[0];
    #pragma unroll
    for (int n = 0; n < GOUT; ++n) acc += tilebuf[wv][lane][n] * head_w[n];
    pred[tile * 16 + lane] = acc;
  }
}

// =====================================================================
extern "C" void kernel_launch(void* const* d_in, const int* in_sizes, int n_in,
                              void* d_out, int out_size, void* d_ws, size_t ws_size,
                              hipStream_t stream) {
  (void)in_sizes; (void)n_in; (void)out_size; (void)ws_size;
  const float* x    = (const float*)d_in[0];
  const float* w2   = (const float*)d_in[1];  const float* b2 = (const float*)d_in[2];
  const float* w3   = (const float*)d_in[3];  const float* b3 = (const float*)d_in[4];
  const float* w5   = (const float*)d_in[5];  const float* b5 = (const float*)d_in[6];
  const float* w7   = (const float*)d_in[7];  const float* b7 = (const float*)d_in[8];
  const float* enc_w = (const float*)d_in[9];  const float* enc_b = (const float*)d_in[10];
  const float* dec_w = (const float*)d_in[11]; const float* dec_b = (const float*)d_in[12];
  const float* semb  = (const float*)d_in[13];
  const float* g0w = (const float*)d_in[14];
  const float* a0s = (const float*)d_in[15]; const float* a0d = (const float*)d_in[16];
  const float* g1w = (const float*)d_in[17];
  const float* a1s = (const float*)d_in[18]; const float* a1d = (const float*)d_in[19];
  const float* pw  = (const float*)d_in[20]; const float* pb  = (const float*)d_in[21];
  const float* hw  = (const float*)d_in[22]; const float* hb  = (const float*)d_in[23];

  float* out = (float*)d_out;

  // ---- workspace carving (offsets kept 256B aligned) ----
  char* ws = (char*)d_ws;
  float*     nf       = (float*)(ws);                       // 16384*64 f32   = 4,194,304 B
  float*     z        = (float*)(ws + 4194304);             // 16384*8  f32   =   524,288
  _Float16*  gi       = (_Float16*)(ws + 4718592);          // 16384*32 f16   = 1,048,576
  float*     h        = (float*)(ws + 5767168);             // 16384*32 f32   = 2,097,152
  _Float16*  hp       = (_Float16*)(ws + 7864320);          // 16384*32 f16   = 1,048,576
  float*     es       = (float*)(ws + 8912896);             // 2*16384 f32    =   131,072
  float*     ed       = (float*)(ws + 9043968);             // 2*16384 f32    =   131,072
  float*     alpha    = (float*)(ws + 9175040);             // 2*16384*24 f32 = 3,145,728
  int*       nbr      = (int*)(ws + 12320768);              // 512*24 int     =    49,152
  int*       cnt      = (int*)(ws + 12369920);              // 512 int        =     2,048
  _Float16*  WcatT    = (_Float16*)(ws + 12371968);         // 32*32 f16      =     2,048
  _Float16*  projT    = (_Float16*)(ws + 12374016);         // 32*32 f16      =     2,048
  float*     rho_part = (float*)(ws + 12376064);            // 64*8 f32       =     2,048

  k_mstcn<<<ROWS / 8, 256, 0, stream>>>(x, w2, b2, w3, b3, w5, b5, w7, b7, nf);
  k_encode<<<ROWS / 256, 256, 0, stream>>>(nf, enc_w, enc_b, semb, z, gi,
                                           out + OUT_SDEV, rho_part);
  k_kl<<<1, 32, 0, stream>>>(rho_part, out + OUT_KL);
  k_recon<<<ROWS, WW, 0, stream>>>(z, dec_w, dec_b, out + OUT_RECON);
  k_prep<<<4, 256, 0, stream>>>(g0w, g1w, pw, WcatT, projT);
  k_graph<<<NN, 32, 0, stream>>>(semb, nbr, cnt);
  k_gat_h<<<ROWS / 16 / 8, 256, 0, stream>>>(gi, WcatT, h);
  k_edges<<<ROWS / 256, 256, 0, stream>>>(h, a0s, a0d, a1s, a1d, es, ed);
  k_stats<<<2 * ROWS / 256, 256, 0, stream>>>(es, ed, nbr, cnt, alpha);
  k_aggr<<<32 * ROWS / 256, 256, 0, stream>>>(h, alpha, nbr, cnt, hp);
  k_proj<<<ROWS / 16 / 8, 256, 0, stream>>>(hp, projT, pb, hw, hb, out + OUT_PRED);
}
